// MSAColumnGlobalAttention_40123584479677
// MI455X (gfx1250) — matrix operand order repaired and checked by reading, not compiled, not run on hardware
//
#include <hip/hip_runtime.h>
#include <hip/hip_bf16.h>
#include <math.h>

#define S_DIM 2048
#define I_DIM 512
#define C_DIM 64
#define XS    72   // padded LDS stride (halves) for 16x64 f16 tiles (144B rows, 16B aligned)
#define WQS   72   // padded LDS stride for transposed weights

typedef __attribute__((ext_vector_type(16))) _Float16 v16h;
typedef __attribute__((ext_vector_type(8)))  _Float16 h8;
typedef __attribute__((ext_vector_type(2)))  _Float16 h2;
typedef __attribute__((ext_vector_type(8)))  float    v8f;

union V16H { v16h v; h8 h[2]; };

// A-matrix 16x32 f16 fragment (M = lane&15; K chunk base = 8*(lane>>4); chunks at +0 and +16)
__device__ __forceinline__ v16h ld_a_frag(const _Float16* base, int stride, int lane, int kOff) {
  const int m  = lane & 15;
  const int kb = (lane >> 4) * 8;
  const _Float16* p = base + m * stride + kOff + kb;
  V16H u;
  u.h[0] = *(const h8*)(p);
  u.h[1] = *(const h8*)(p + 16);
  return u.v;
}

// B-matrix 32x16 f16 fragment from N-major (transposed) weights:
// N = lane&15 (+16*nTile), K chunk base = kOff + 8*(lane>>4); chunks at +0 and +16
__device__ __forceinline__ v16h ld_b_frag(const _Float16* base, int stride, int lane, int nTile, int kOff) {
  const int n  = lane & 15;
  const int kb = (lane >> 4) * 8;
  const _Float16* p = base + (nTile * 16 + n) * stride + kOff + kb;
  V16H u;
  u.h[0] = *(const h8*)(p);
  u.h[1] = *(const h8*)(p + 16);
  return u.v;
}

// LayerNorm a 16x64 f32 LDS tile into a 16xXS f16 LDS tile.
// Two lanes per row: lane handles 32 elements, partner (lane^16) the other 32.
__device__ __forceinline__ void ln_tile(const float* __restrict__ rawTile,
                                        _Float16* __restrict__ xTile,
                                        const float* __restrict__ gam,
                                        const float* __restrict__ bet,
                                        int lane) {
  const int m    = lane & 15;
  const int half = lane >> 4;           // which 32-element half this lane owns
  const float* rw = rawTile + m * 64 + half * 32;
  float s = 0.f, ss = 0.f;
#pragma unroll
  for (int c = 0; c < 32; c += 4) {
    const float4 v = *(const float4*)(rw + c);
    s  += v.x + v.y + v.z + v.w;
    ss += v.x * v.x + v.y * v.y + v.z * v.z + v.w * v.w;
  }
  s  += __shfl_xor(s, 16, 32);
  ss += __shfl_xor(ss, 16, 32);
  const float mu   = s * (1.f / 64.f);
  const float var  = ss * (1.f / 64.f) - mu * mu;
  const float rinv = rsqrtf(var + 1e-5f);
  _Float16* xr   = xTile + m * XS + half * 32;
  const float* g = gam + half * 32;
  const float* b = bet + half * 32;
#pragma unroll
  for (int c = 0; c < 32; c += 2) {
    h2 p;
    p.x = (_Float16)((rw[c]     - mu) * rinv * g[c]     + b[c]);
    p.y = (_Float16)((rw[c + 1] - mu) * rinv * g[c + 1] + b[c + 1]);
    *(h2*)(xr + c) = p;   // 4B-aligned packed store
  }
}

// ---------------- Kernel 1: LayerNorm + QKV projection + q accumulation ----------------
// grid (I/16, S/64), block 128 (4 waves). Wave w handles t = by*64 + w*16 .. +15, i tile = bx*16.
__global__ __launch_bounds__(128)
void k1_ln_qkv(const float* __restrict__ msa,
               const float* __restrict__ gamma_,
               const float* __restrict__ beta_,
               const float* __restrict__ wqkv,
               float* __restrict__ kbuf,   // (I, S, 8)
               float* __restrict__ vbuf,   // (I, S, 8)
               float* __restrict__ qsum)   // (I, 64)
{
  __shared__ __align__(16) float    raw[4][16 * 64];
  __shared__ __align__(16) _Float16 xt[4][16 * XS];
  __shared__ __align__(16) _Float16 wt[80 * WQS];
  __shared__ float gam[64], bet[64];

  const int tid   = threadIdx.x;
  const int wave  = tid >> 5;
  const int lane  = tid & 31;
  const int iBase = blockIdx.x * 16;
  const int tBase = blockIdx.y * 64 + wave * 16;
  const int hi    = lane >> 4;
  const int nn    = lane & 15;

  // transpose W_qkv (64x80 f32) -> wt[n][c] f16
  for (int idx = tid; idx < 64 * 80; idx += 128) {
    const int c = idx / 80, n = idx % 80;
    wt[n * WQS + c] = (_Float16)wqkv[idx];
  }
  if (tid < 64) { gam[tid] = gamma_[tid]; bet[tid] = beta_[tid]; }
  __syncthreads();

  float qacc[4][8];
#pragma unroll
  for (int n = 0; n < 4; ++n)
#pragma unroll
    for (int r = 0; r < 8; ++r) qacc[n][r] = 0.f;

  // branchless k/v destination: lanes with N<8 write k channel N, others v channel N-8
  float* kvdst = (nn < 8) ? (kbuf + nn) : (vbuf + (nn - 8));

  for (int j = 0; j < 16; ++j) {
    const int t = tBase + j;
    const size_t rowBase = (size_t)t * I_DIM + iBase;
    const float* src = msa + rowBase * C_DIM;
    // cooperative 4KB contiguous tile load
#pragma unroll
    for (int chk = 0; chk < 8; ++chk) {
      const int o = chk * 128 + lane * 4;
      *(float4*)&raw[wave][o] = *(const float4*)(src + o);
    }
    if (j + 1 < 16) __builtin_prefetch(src + (size_t)I_DIM * C_DIM + lane * 32, 0, 1);
    __syncthreads();

    ln_tile(raw[wave], xt[wave], gam, bet, lane);
    __syncthreads();

    const v16h a0 = ld_a_frag(xt[wave], XS, lane, 0);
    const v16h a1 = ld_a_frag(xt[wave], XS, lane, 32);
#pragma unroll
    for (int n = 0; n < 5; ++n) {
      const v16h b0 = ld_b_frag(wt, WQS, lane, n, 0);
      const v16h b1 = ld_b_frag(wt, WQS, lane, n, 32);
      v8f c = {};
      c = __builtin_amdgcn_wmma_f32_16x16x32_f16(false, a0, false, b0, (short)0, c, false, false);
      c = __builtin_amdgcn_wmma_f32_16x16x32_f16(false, a1, false, b1, (short)0, c, false, false);
      if (n < 4) {
#pragma unroll
        for (int r = 0; r < 8; ++r) qacc[n][r] += c[r];
      } else {
        const size_t base = (((size_t)(iBase + 8 * hi)) * S_DIM + (size_t)t) * 8;
#pragma unroll
        for (int r = 0; r < 8; ++r)
          kvdst[base + (size_t)r * (S_DIM * 8)] = c[r];  // unconditional stores
      }
    }
    __syncthreads();  // protect raw/xt before next iteration
  }

#pragma unroll
  for (int n = 0; n < 4; ++n)
#pragma unroll
    for (int r = 0; r < 8; ++r) {
      const int i_r = iBase + r + 8 * hi;
      atomicAdd(&qsum[i_r * 64 + n * 16 + nn], qacc[n][r]);
    }
}

// ---------------- Kernel 3: flash softmax over t + context ----------------
// grid (I), block 256 = 8 waves, wave = head h; lane strides over t.
__global__ __launch_bounds__(256)
void k3_attn(const float* __restrict__ kbuf,
             const float* __restrict__ vbuf,
             const float* __restrict__ qsum,
             float* __restrict__ ctx)   // (I, 64) as [i*64 + d*8 + h]
{
  const int i    = blockIdx.x;
  const int h    = threadIdx.x >> 5;
  const int lane = threadIdx.x & 31;
  const float scl = 0.35355339059327373f * (1.f / 2048.f);  // (1/sqrt(8)) * (1/S)

  float qm[8];
#pragma unroll
  for (int d = 0; d < 8; ++d) qm[d] = qsum[i * 64 + d * 8 + h] * scl;

  const float* kp = kbuf + (size_t)i * S_DIM * 8;
  const float* vp = vbuf + (size_t)i * S_DIM * 8;

  float m = -INFINITY, s = 0.f;
  float acc[8];
#pragma unroll
  for (int d = 0; d < 8; ++d) acc[d] = 0.f;

  for (int it = 0; it < S_DIM / 32; ++it) {
    const size_t t = (size_t)it * 32 + lane;
    const float4 k0 = *(const float4*)(kp + t * 8);
    const float4 k1 = *(const float4*)(kp + t * 8 + 4);
    const float4 v0 = *(const float4*)(vp + t * 8);
    const float4 v1 = *(const float4*)(vp + t * 8 + 4);
    const float logit = qm[0] * k0.x + qm[1] * k0.y + qm[2] * k0.z + qm[3] * k0.w
                      + qm[4] * k1.x + qm[5] * k1.y + qm[6] * k1.z + qm[7] * k1.w;
    const float nm = fmaxf(m, logit);
    const float f  = __expf(m - nm);
    const float w  = __expf(logit - nm);
    s = s * f + w;
    const float vv[8] = {v0.x, v0.y, v0.z, v0.w, v1.x, v1.y, v1.z, v1.w};
#pragma unroll
    for (int d = 0; d < 8; ++d) acc[d] = acc[d] * f + w * vv[d];
    m = nm;
  }
  // cross-lane flash combine (wave32)
#pragma unroll
  for (int off = 16; off > 0; off >>= 1) {
    const float m2 = __shfl_xor(m, off, 32);
    const float s2 = __shfl_xor(s, off, 32);
    const float nm = fmaxf(m, m2);
    const float f1 = __expf(m - nm);
    const float f2 = __expf(m2 - nm);
    s = s * f1 + s2 * f2;
#pragma unroll
    for (int d = 0; d < 8; ++d) {
      const float a2 = __shfl_xor(acc[d], off, 32);
      acc[d] = acc[d] * f1 + a2 * f2;
    }
    m = nm;
  }
  if (lane == 0) {
    const float inv = 1.f / s;
#pragma unroll
    for (int d = 0; d < 8; ++d) ctx[i * 64 + d * 8 + h] = acc[d] * inv;
  }
}

// ---------------- Kernel 4: LN + gate GEMM + sigmoid*ctx + output GEMM ----------------
// grid (I/16, S/4), block 128; wave handles t = by*4+wave, i tile = bx*16.
__global__ __launch_bounds__(128)
void k4_out(const float* __restrict__ msa,
            const float* __restrict__ gamma_,
            const float* __restrict__ beta_,
            const float* __restrict__ wg,
            const float* __restrict__ bg,
            const float* __restrict__ wo,
            const float* __restrict__ bo,
            const float* __restrict__ ctx,
            float* __restrict__ out)
{
  __shared__ __align__(16) float    raw[4][16 * 64];
  __shared__ __align__(16) _Float16 xt[4][16 * XS];
  __shared__ __align__(16) _Float16 yt[4][16 * XS];
  __shared__ __align__(16) _Float16 wgt[64 * WQS];
  __shared__ __align__(16) _Float16 wot[64 * WQS];
  __shared__ float gam[64], bet[64], bgs[64], bos[64];

  const int tid   = threadIdx.x;
  const int wave  = tid >> 5;
  const int lane  = tid & 31;
  const int iBase = blockIdx.x * 16;
  const int t     = blockIdx.y * 4 + wave;
  const int hi    = lane >> 4;
  const int nn    = lane & 15;

  for (int idx = tid; idx < 64 * 64; idx += 128) {
    const int c = idx >> 6, n = idx & 63;
    wgt[n * WQS + c] = (_Float16)wg[idx];
    wot[n * WQS + c] = (_Float16)wo[idx];
  }
  if (tid < 64) { gam[tid] = gamma_[tid]; bet[tid] = beta_[tid]; bgs[tid] = bg[tid]; bos[tid] = bo[tid]; }
  __syncthreads();

  const size_t rowBase = (size_t)t * I_DIM + iBase;
  const float* src = msa + rowBase * C_DIM;
#pragma unroll
  for (int chk = 0; chk < 8; ++chk) {
    const int o = chk * 128 + lane * 4;
    *(float4*)&raw[wave][o] = *(const float4*)(src + o);
  }
  __syncthreads();

  ln_tile(raw[wave], xt[wave], gam, bet, lane);
  __syncthreads();

  // gate projection: g = sigmoid(x @ Wg + bg); y = g * ctx  -> yt (f16)
  const v16h a0 = ld_a_frag(xt[wave], XS, lane, 0);
  const v16h a1 = ld_a_frag(xt[wave], XS, lane, 32);
#pragma unroll
  for (int n = 0; n < 4; ++n) {
    const v16h b0 = ld_b_frag(wgt, WQS, lane, n, 0);
    const v16h b1 = ld_b_frag(wgt, WQS, lane, n, 32);
    v8f c = {};
    c = __builtin_amdgcn_wmma_f32_16x16x32_f16(false, a0, false, b0, (short)0, c, false, false);
    c = __builtin_amdgcn_wmma_f32_16x16x32_f16(false, a1, false, b1, (short)0, c, false, false);
#pragma unroll
    for (int r = 0; r < 8; ++r) {
      const int chn = n * 16 + nn;
      const int M   = r + 8 * hi;
      const int i_r = iBase + M;
      const float gv = 1.f / (1.f + __expf(-(c[r] + bgs[chn])));
      yt[wave][M * XS + chn] = (_Float16)(gv * ctx[i_r * 64 + chn]);
    }
  }
  __syncthreads();

  // output projection: out = y @ Wo + bo
  const v16h y0 = ld_a_frag(yt[wave], XS, lane, 0);
  const v16h y1 = ld_a_frag(yt[wave], XS, lane, 32);
#pragma unroll
  for (int n = 0; n < 4; ++n) {
    const v16h b0 = ld_b_frag(wot, WQS, lane, n, 0);
    const v16h b1 = ld_b_frag(wot, WQS, lane, n, 32);
    v8f c = {};
    c = __builtin_amdgcn_wmma_f32_16x16x32_f16(false, y0, false, b0, (short)0, c, false, false);
    c = __builtin_amdgcn_wmma_f32_16x16x32_f16(false, y1, false, b1, (short)0, c, false, false);
#pragma unroll
    for (int r = 0; r < 8; ++r) {
      const int M = r + 8 * hi;
      out[(rowBase + M) * 64 + n * 16 + nn] = c[r] + bos[n * 16 + nn];
    }
  }
}

extern "C" void kernel_launch(void* const* d_in, const int* in_sizes, int n_in,
                              void* d_out, int out_size, void* d_ws, size_t ws_size,
                              hipStream_t stream) {
  const float* msa   = (const float*)d_in[0];
  const float* gamma = (const float*)d_in[1];
  const float* beta  = (const float*)d_in[2];
  const float* wqkv  = (const float*)d_in[3];
  const float* wg    = (const float*)d_in[4];
  const float* bg    = (const float*)d_in[5];
  const float* wo    = (const float*)d_in[6];
  const float* bo    = (const float*)d_in[7];
  float* out = (float*)d_out;

  // workspace: kbuf (32MB) | vbuf (32MB) | qsum (128KB) | ctx (128KB)
  char* ws = (char*)d_ws;
  const size_t kvBytes = (size_t)S_DIM * I_DIM * 8 * sizeof(float);
  float* kbuf = (float*)ws;
  float* vbuf = (float*)(ws + kvBytes);
  float* qsum = (float*)(ws + 2 * kvBytes);
  float* ctxb = qsum + I_DIM * 64;

  hipMemsetAsync(qsum, 0, (size_t)I_DIM * 64 * sizeof(float), stream);
  k1_ln_qkv<<<dim3(I_DIM / 16, S_DIM / 64), 128, 0, stream>>>(msa, gamma, beta, wqkv, kbuf, vbuf, qsum);
  k3_attn<<<dim3(I_DIM), 256, 0, stream>>>(kbuf, vbuf, qsum, ctxb);
  k4_out<<<dim3(I_DIM / 16, S_DIM / 4), 128, 0, stream>>>(msa, gamma, beta, wg, bg, wo, bo, ctxb, out);
}
